// MixPhonemeRobertaLMHead_25211458027708
// MI455X (gfx1250) — matrix operand order repaired and checked by reading, not compiled, not run on hardware
//
#include <hip/hip_runtime.h>
#include <hip/hip_bf16.h>
#include <math.h>

// CDNA5 / gfx1250: wave32, WMMA 16x16x32 bf16 with f32 accumulate.
typedef __attribute__((ext_vector_type(16))) __bf16 v16bf;
typedef __attribute__((ext_vector_type(8)))  float  v8f;

#define LN_EPS 1e-5f

__device__ __forceinline__ unsigned short f2bf(float f) {
    union { float f; unsigned u; } c; c.f = f;
    unsigned u = c.u;
    u += 0x7FFFu + ((u >> 16) & 1u);   // round-to-nearest-even
    return (unsigned short)(u >> 16);
}

// ---------------------------------------------------------------------------
// Kernel 1: per-block count of segment-start flags (flag[i] = t[i] != t[i-1])
// ---------------------------------------------------------------------------
__global__ void k_flags(const int* __restrict__ tgt, int* __restrict__ blockSums, int N) {
    int t = threadIdx.x;
    int i = blockIdx.x * 256 + t;
    int flag = 0;
    if (i < N) flag = (i == 0) || (tgt[i] != tgt[i - 1]);
    __shared__ int sd[256];
    sd[t] = flag; __syncthreads();
    for (int off = 1; off < 256; off <<= 1) {
        int x = (t >= off) ? sd[t - off] : 0;
        __syncthreads();
        sd[t] += x;
        __syncthreads();
    }
    if (t == 255) blockSums[blockIdx.x] = sd[255];
}

// ---------------------------------------------------------------------------
// Kernel 2: single-block exclusive scan of block sums; also seg_start[S] = N
// ---------------------------------------------------------------------------
__global__ void k_scan_blocks(const int* __restrict__ blockSums, int* __restrict__ blockOff,
                              int nb, const int* __restrict__ numWordsPtr,
                              int* __restrict__ segStart, int N) {
    int t = threadIdx.x;
    __shared__ int sd[256];
    int v = (t < nb) ? blockSums[t] : 0;
    sd[t] = v; __syncthreads();
    for (int off = 1; off < 256; off <<= 1) {
        int x = (t >= off) ? sd[t - off] : 0;
        __syncthreads();
        sd[t] += x;
        __syncthreads();
    }
    if (t < nb) blockOff[t] = sd[t] - v;       // exclusive prefix
    if (t == 0) segStart[numWordsPtr[0]] = N;  // sentinel end
}

// ---------------------------------------------------------------------------
// Kernel 3: recompute flags, intra-block scan, scatter segment start indices
// ---------------------------------------------------------------------------
__global__ void k_seg_starts(const int* __restrict__ tgt, const int* __restrict__ blockOff,
                             int* __restrict__ segStart, int N) {
    int t = threadIdx.x;
    int i = blockIdx.x * 256 + t;
    int flag = 0;
    if (i < N) flag = (i == 0) || (tgt[i] != tgt[i - 1]);
    __shared__ int sd[256];
    sd[t] = flag; __syncthreads();
    for (int off = 1; off < 256; off <<= 1) {
        int x = (t >= off) ? sd[t - off] : 0;
        __syncthreads();
        sd[t] += x;
        __syncthreads();
    }
    if (flag) segStart[blockOff[blockIdx.x] + sd[t] - 1] = i;
}

// ---------------------------------------------------------------------------
// Kernel 4: per-segment mean pool -> bf16 activations, + seg_targets to tail
// ---------------------------------------------------------------------------
__global__ __launch_bounds__(256)
void k_pool(const float* __restrict__ feat, const int* __restrict__ tgt,
            const int* __restrict__ segStart, unsigned short* __restrict__ pooledBf,
            float* __restrict__ outTail) {
    const int s = blockIdx.x, t = threadIdx.x;
    const int start = segStart[s], end = segStart[s + 1];
    float a0 = 0.f, a1 = 0.f, a2 = 0.f;
    const float* p = feat + (size_t)start * 768;
    for (int tok = start; tok < end; ++tok, p += 768) {
        a0 += p[t];
        a1 += p[t + 256];
        a2 += p[t + 512];
    }
    const float inv = 1.0f / (float)(end - start);
    unsigned short* q = pooledBf + (size_t)s * 768;
    q[t]       = f2bf(a0 * inv);
    q[t + 256] = f2bf(a1 * inv);
    q[t + 512] = f2bf(a2 * inv);
    if (t == 0) outTail[s] = (float)tgt[start];
}

// ---------------------------------------------------------------------------
// Kernel 5: bulk f32 -> bf16 conversion (weights), 4 elems/thread
// ---------------------------------------------------------------------------
__global__ __launch_bounds__(256)
void k_f32_to_bf16(const float* __restrict__ src, unsigned short* __restrict__ dst, int n4) {
    int i = blockIdx.x * 256 + threadIdx.x;
    if (i < n4) {
        float4 f = ((const float4*)src)[i];
        uint2 p;
        p.x = (unsigned)f2bf(f.x) | ((unsigned)f2bf(f.y) << 16);
        p.y = (unsigned)f2bf(f.z) | ((unsigned)f2bf(f.w) << 16);
        ((uint2*)dst)[i] = p;
    }
}

// ---------------------------------------------------------------------------
// Kernel 6: tiled WMMA GEMM: C[M x Ncols] = A[M x K] * B[Ncols x K]^T + bias
//   A, B bf16 (K-major). WG tile 128(M) x 256(N) x 32(K), 256 threads =
//   8 wave32 in a 2(M) x 4(N) grid; each wave owns a 64x64 tile = 4x4
//   16x16 fragments -> 16 WMMAs per K-step per wave, 8 fragment loads.
//   Double-buffered LDS (80B row stride: 16B-aligned, conflict-free) with a
//   register pipeline: next tile's global loads issue before current tile's
//   ds_load/WMMA work; one barrier per K-step.
// ---------------------------------------------------------------------------
#define LDSK 40  // shorts per LDS row (32 payload + 8 pad) -> 80 bytes

__global__ __launch_bounds__(256)
void k_gemm_bf16(const unsigned short* __restrict__ A, const unsigned short* __restrict__ B,
                 const float* __restrict__ bias, float* __restrict__ C,
                 int K, int Ncols) {
    __shared__ unsigned short As[2][128 * LDSK];
    __shared__ unsigned short Bs[2][256 * LDSK];

    const int tid  = threadIdx.x;
    const int lane = tid & 31;
    const int wid  = tid >> 5;
    const int waveMoff = (wid & 1) * 64;
    const int waveNoff = (wid >> 1) * 64;
    const int mBase = blockIdx.y * 128;
    const int nBase = blockIdx.x * 256;
    const int r  = lane & 15;   // row/col within fragment
    const int kh = lane >> 4;   // lane half selects K grouping

    // Staging coords: A tile 128x32 = 512 uint4 (2/thread),
    //                 B tile 256x32 = 1024 uint4 (4/thread).
    const int srow = tid >> 2;           // rows 0..63 (+64 / +128 / +192)
    const int scol = (tid & 3) * 8;      // bf16 element offset within row

    const v8f vzero = {};
    v8f acc[4][4];
#pragma unroll
    for (int m = 0; m < 4; ++m)
#pragma unroll
        for (int n = 0; n < 4; ++n) acc[m][n] = vzero;

    const unsigned short* aBase = A + (size_t)(mBase + srow) * K + scol;
    const unsigned short* bBase = B + (size_t)(nBase + srow) * K + scol;
    const size_t rowStep = (size_t)64 * K;

    const int nk = K / 32;

    // Prologue: stage tile 0 into buffer 0.
    {
        uint4 ra0 = *(const uint4*)(aBase);
        uint4 ra1 = *(const uint4*)(aBase + rowStep);
        uint4 rb0 = *(const uint4*)(bBase);
        uint4 rb1 = *(const uint4*)(bBase + rowStep);
        uint4 rb2 = *(const uint4*)(bBase + 2 * rowStep);
        uint4 rb3 = *(const uint4*)(bBase + 3 * rowStep);
        *(uint4*)&As[0][srow * LDSK + scol]         = ra0;
        *(uint4*)&As[0][(srow + 64) * LDSK + scol]  = ra1;
        *(uint4*)&Bs[0][srow * LDSK + scol]         = rb0;
        *(uint4*)&Bs[0][(srow + 64) * LDSK + scol]  = rb1;
        *(uint4*)&Bs[0][(srow + 128) * LDSK + scol] = rb2;
        *(uint4*)&Bs[0][(srow + 192) * LDSK + scol] = rb3;
    }
    __syncthreads();

    for (int kt = 0; kt < nk; ++kt) {
        const int buf = kt & 1;
        const bool havenext = (kt + 1) < nk;

        // Issue next tile's global loads first (latency overlapped with WMMA).
        uint4 ra0, ra1, rb0, rb1, rb2, rb3;
        if (havenext) {
            const int kn = (kt + 1) * 32;
            ra0 = *(const uint4*)(aBase + kn);
            ra1 = *(const uint4*)(aBase + kn + rowStep);
            rb0 = *(const uint4*)(bBase + kn);
            rb1 = *(const uint4*)(bBase + kn + rowStep);
            rb2 = *(const uint4*)(bBase + kn + 2 * rowStep);
            rb3 = *(const uint4*)(bBase + kn + 3 * rowStep);
        }
        if (kt + 2 < nk) {  // cache hint for the K+2 tile (global_prefetch_b8)
            __builtin_prefetch(aBase + (kt + 2) * 32, 0, 1);
            __builtin_prefetch(bBase + (kt + 2) * 32, 0, 1);
        }

        // A fragment per ISA 16-bit A 16x32 layout:
        //   lane (r, kh): VGPR 0-3 -> K = kh*8 + 0..7 ; VGPR 4-7 -> K = 16 + kh*8 + 0..7
        v16bf afr[4];
#pragma unroll
        for (int mt = 0; mt < 4; ++mt) {
            const unsigned short* p = &As[buf][(waveMoff + mt * 16 + r) * LDSK];
            union { uint4 u[2]; v16bf v; } fa;
            fa.u[0] = *(const uint4*)(p + kh * 8);
            fa.u[1] = *(const uint4*)(p + 16 + kh * 8);
            afr[mt] = fa.v;
        }
        // B fragment: lane (n, kh) holds K = kh*8 + 0..7 and 16 + kh*8 + 0..7 (matches A)
        v16bf bfr[4];
#pragma unroll
        for (int nt = 0; nt < 4; ++nt) {
            const unsigned short* p = &Bs[buf][(waveNoff + nt * 16 + r) * LDSK + kh * 8];
            union { uint4 u[2]; v16bf v; } fb;
            fb.u[0] = *(const uint4*)(p);
            fb.u[1] = *(const uint4*)(p + 16);
            bfr[nt] = fb.v;
        }
#pragma unroll
        for (int mt = 0; mt < 4; ++mt)
#pragma unroll
            for (int nt = 0; nt < 4; ++nt)
                acc[mt][nt] = __builtin_amdgcn_wmma_f32_16x16x32_bf16(
                    false, afr[mt], false, bfr[nt], (short)0, acc[mt][nt], false, false);

        // Store next tile into the alternate buffer; current readers are done
        // with it (barrier at end of previous iteration guarantees this).
        if (havenext) {
            const int nb_ = buf ^ 1;
            *(uint4*)&As[nb_][srow * LDSK + scol]         = ra0;
            *(uint4*)&As[nb_][(srow + 64) * LDSK + scol]  = ra1;
            *(uint4*)&Bs[nb_][srow * LDSK + scol]         = rb0;
            *(uint4*)&Bs[nb_][(srow + 64) * LDSK + scol]  = rb1;
            *(uint4*)&Bs[nb_][(srow + 128) * LDSK + scol] = rb2;
            *(uint4*)&Bs[nb_][(srow + 192) * LDSK + scol] = rb3;
        }
        __syncthreads();
    }

    // Store: C/D layout -> lane L: N = L%16, rows M = (L/16)*8 + vgpr
#pragma unroll
    for (int nt = 0; nt < 4; ++nt) {
        const int col = nBase + waveNoff + nt * 16 + r;
        const float bv = bias[col];
#pragma unroll
        for (int mt = 0; mt < 4; ++mt) {
            const int row0 = mBase + waveMoff + mt * 16 + kh * 8;
            float* cp = C + (size_t)row0 * Ncols + col;
            v8f a = acc[mt][nt];
#pragma unroll
            for (int rr = 0; rr < 8; ++rr)
                cp[(size_t)rr * Ncols] = a[rr] + bv;
        }
    }
}

// ---------------------------------------------------------------------------
// Kernel 7: fused exact GELU + LayerNorm per row (768 = 3 * 256) -> bf16 out
// ---------------------------------------------------------------------------
__global__ __launch_bounds__(256)
void k_gelu_ln(const float* __restrict__ X, const float* __restrict__ gamma,
               const float* __restrict__ beta, unsigned short* __restrict__ Ybf) {
    const int srow = blockIdx.x, t = threadIdx.x;
    const float* x = X + (size_t)srow * 768;
    const float v0 = x[t], v1 = x[t + 256], v2 = x[t + 512];
    const float g0 = 0.5f * v0 * (1.0f + erff(v0 * 0.70710678118654752f));
    const float g1 = 0.5f * v1 * (1.0f + erff(v1 * 0.70710678118654752f));
    const float g2 = 0.5f * v2 * (1.0f + erff(v2 * 0.70710678118654752f));

    __shared__ float s1[256], s2[256];
    s1[t] = g0 + g1 + g2;
    s2[t] = g0 * g0 + g1 * g1 + g2 * g2;
    __syncthreads();
    for (int off = 128; off > 0; off >>= 1) {
        if (t < off) { s1[t] += s1[t + off]; s2[t] += s2[t + off]; }
        __syncthreads();
    }
    const float mu   = s1[0] * (1.0f / 768.0f);
    const float var  = s2[0] * (1.0f / 768.0f) - mu * mu;
    const float rstd = rsqrtf(var + LN_EPS);

    unsigned short* y = Ybf + (size_t)srow * 768;
    y[t]       = f2bf((g0 - mu) * rstd * gamma[t]       + beta[t]);
    y[t + 256] = f2bf((g1 - mu) * rstd * gamma[t + 256] + beta[t + 256]);
    y[t + 512] = f2bf((g2 - mu) * rstd * gamma[t + 512] + beta[t + 512]);
}

// ---------------------------------------------------------------------------
// Host-side launcher
// ---------------------------------------------------------------------------
extern "C" void kernel_launch(void* const* d_in, const int* in_sizes, int n_in,
                              void* d_out, int out_size, void* d_ws, size_t ws_size,
                              hipStream_t stream) {
    const float* features = (const float*)d_in[0];
    const int*   targets  = (const int*)  d_in[1];
    const int*   numWords = (const int*)  d_in[2];
    const float* Wd       = (const float*)d_in[3];
    const float* bd       = (const float*)d_in[4];
    const float* gamma    = (const float*)d_in[5];
    const float* beta     = (const float*)d_in[6];
    const float* Wv       = (const float*)d_in[7];
    const float* bv       = (const float*)d_in[8];
    float* out = (float*)d_out;

    const int D = 768;
    const int N = in_sizes[0] / D;          // 65536 tokens
    const int V = in_sizes[7] / D;          // 32000 vocab
    const int S = out_size / (V + 1);       // 8192 segments (logits + targets)
    const int nb = (N + 255) / 256;         // 256 scan blocks

    // Workspace layout (256B-aligned slices)
    char* ws = (char*)d_ws;
    int* blockSums = (int*)(ws);
    int* blockOff  = (int*)(ws + 4096);
    int* segStart  = (int*)(ws + 8192);                    // S+1 ints
    size_t off = 8192 + ((((size_t)(S + 1) * 4) + 255) & ~(size_t)255);
    auto take = [&](size_t bytes) {
        char* p = ws + off;
        off += (bytes + 255) & ~(size_t)255;
        return p;
    };
    unsigned short* pooledBf = (unsigned short*)take((size_t)S * D * 2); // bf16 activations
    unsigned short* WdBf     = (unsigned short*)take((size_t)D * D * 2); // bf16 W_dense
    unsigned short* WvBf     = (unsigned short*)take((size_t)V * D * 2); // bf16 W_vocab
    float*          dense    = (float*)take((size_t)S * D * 4);          // f32 dense out
    unsigned short* xlnBf    = (unsigned short*)take((size_t)S * D * 2); // bf16 LN out
    float* outTail = out + (size_t)S * V;   // seg_targets as float

    // 1-3: segment boundaries via block scan
    k_flags      <<<nb, 256, 0, stream>>>(targets, blockSums, N);
    k_scan_blocks<<<1, 256, 0, stream>>>(blockSums, blockOff, nb, numWords, segStart, N);
    k_seg_starts <<<nb, 256, 0, stream>>>(targets, blockOff, segStart, N);

    // 4: mean pool per segment -> bf16 (+ seg_targets to output tail)
    k_pool<<<S, 256, 0, stream>>>(features, targets, segStart, pooledBf, outTail);

    // 5: one-shot weight conversions to bf16
    k_f32_to_bf16<<<(D * D / 4 + 255) / 256, 256, 0, stream>>>(Wd, WdBf, D * D / 4);
    k_f32_to_bf16<<<(V * D / 4 + 255) / 256, 256, 0, stream>>>(Wv, WvBf, V * D / 4);

    // 6: dense 768x768 projection (bf16 WMMA, f32 accum)
    dim3 gd(D / 256, S / 128);   // 3 x 64
    k_gemm_bf16<<<gd, 256, 0, stream>>>(pooledBf, WdBf, bd, dense, D, D);

    // 7: GELU + LayerNorm -> bf16
    k_gelu_ln<<<S, 256, 0, stream>>>(dense, gamma, beta, xlnBf);

    // 8: vocab projection 8192 x 32000 x 768 (bf16 WMMA, f32 accum)
    dim3 gv(V / 256, S / 128);   // 125 x 64
    k_gemm_bf16<<<gv, 256, 0, stream>>>(xlnBf, WvBf, bv, out, D, V);
}
